// Net_72438918414797
// MI455X (gfx1250) — matrix operand order repaired and checked by reading, not compiled, not run on hardware
//
#include <hip/hip_runtime.h>

typedef _Float16 v16h __attribute__((ext_vector_type(16)));
typedef _Float16 v8h  __attribute__((ext_vector_type(8)));
typedef float    v8f  __attribute__((ext_vector_type(8)));

union V16 { v16h v; v8h h[2]; };

__device__ __forceinline__ v8f wmma32(const V16& a, const V16& b, v8f c) {
  return __builtin_amdgcn_wmma_f32_16x16x32_f16(false, a.v, false, b.v, (short)0, c,
                                                false, false);
}

// ---------------------------------------------------------------- utilities
__global__ void k_zero(float* __restrict__ p, long long n) {
  long long i = (long long)blockIdx.x * blockDim.x + threadIdx.x;
  if (i < n) p[i] = 0.f;
}

__global__ void k_rowinit(int* __restrict__ rowinfo, int n) {
  int i = blockIdx.x * blockDim.x + threadIdx.x;
  if (i < n) rowinfo[i] = -1;
}

__global__ void k_rowscatter(int* __restrict__ rowinfo, const int* __restrict__ av,
                             const int* __restrict__ ac, int nVar, int nCon) {
  int t = blockIdx.x * blockDim.x + threadIdx.x;
  if (t < nVar) rowinfo[av[t]] = t;
  else if (t < nVar + nCon) rowinfo[ac[t - nVar]] = nVar + (t - nVar);
}

// Build f16 transposed + zero-padded weights.
// w1t/wct/wvt: per layer [n][k] 64x64.  fc1t: [n][k] 64x320.  fc2t/fc3t: 64x64.
__global__ void k_wprep(const float* __restrict__ h2vw1, const float* __restrict__ wcons,
                        const float* __restrict__ wvars, const float* __restrict__ fc1w,
                        const float* __restrict__ fc2w, const float* __restrict__ fc3w,
                        _Float16* __restrict__ w1t, _Float16* __restrict__ wct,
                        _Float16* __restrict__ wvt, _Float16* __restrict__ fc1t,
                        _Float16* __restrict__ fc2t, _Float16* __restrict__ fc3t, int L) {
  int gid = blockIdx.x * blockDim.x + threadIdx.x;
  int per = L * 4096;
  if (gid < per) {                       // w1t: src h2v_w1 (L,64,63)
    int l = gid >> 12, n = (gid >> 6) & 63, k = gid & 63;
    float v = (n < 63) ? h2vw1[(l * 64 + k) * 63 + n] : 0.f;
    w1t[gid] = (_Float16)v; return;
  }
  gid -= per;
  if (gid < per) {                       // wct: src w_cons (L,63,63)
    int l = gid >> 12, n = (gid >> 6) & 63, k = gid & 63;
    float v = (n < 63 && k < 63) ? wcons[(l * 63 + k) * 63 + n] : 0.f;
    wct[gid] = (_Float16)v; return;
  }
  gid -= per;
  if (gid < per) {                       // wvt: src w_vars (L,63,63)
    int l = gid >> 12, n = (gid >> 6) & 63, k = gid & 63;
    float v = (n < 63 && k < 63) ? wvars[(l * 63 + k) * 63 + n] : 0.f;
    wvt[gid] = (_Float16)v; return;
  }
  gid -= per;
  if (gid < 64 * 320) {                  // fc1t: src fc1_w (320,64)
    int n = gid / 320, k = gid % 320;
    fc1t[gid] = (_Float16)fc1w[k * 64 + n]; return;
  }
  gid -= 64 * 320;
  if (gid < 4096) { int n = gid >> 6, k = gid & 63; fc2t[gid] = (_Float16)fc2w[k * 64 + n]; return; }
  gid -= 4096;
  if (gid < 4096) { int n = gid >> 6, k = gid & 63; fc3t[gid] = (_Float16)fc3w[k * 64 + n]; return; }
}

// ---------------------------------------------------------------- embedding
__global__ void k_embed(const float* __restrict__ vnf, const float* __restrict__ cnf,
                        const float* __restrict__ vw1, const float* __restrict__ vb1,
                        const float* __restrict__ vw2, const float* __restrict__ vb2,
                        const float* __restrict__ cw1, const float* __restrict__ cb1,
                        const float* __restrict__ cw2, const float* __restrict__ cb2,
                        const int* __restrict__ av, const int* __restrict__ ac,
                        float* __restrict__ x32, _Float16* __restrict__ xh,
                        _Float16* __restrict__ hcath, int nVar, int nCon) {
  int t = blockIdx.x * blockDim.x + threadIdx.x;
  if (t >= nVar + nCon) return;
  bool isVar = t < nVar;
  int i = isVar ? t : t - nVar;
  const float* feat = isVar ? vnf : cnf;
  const float* w1p = isVar ? vw1 : cw1;
  const float* b1p = isVar ? vb1 : cb1;
  const float* w2p = isVar ? vw2 : cw2;
  const float* b2p = isVar ? vb2 : cb2;
  float f0 = feat[2 * i], f1 = feat[2 * i + 1];
  float o[61];
#pragma unroll
  for (int j = 0; j < 61; ++j) o[j] = b2p[j];
  for (int k = 0; k < 61; ++k) {
    float hk = fmaxf(f0 * w1p[k] + f1 * w1p[61 + k] + b1p[k], 0.f);
#pragma unroll
    for (int j = 0; j < 61; ++j) o[j] += hk * w2p[k * 61 + j];
  }
  int dest = isVar ? av[i] : ac[i];
  float* xr = x32 + (size_t)dest * 64;
  _Float16* hr = xh + (size_t)dest * 64;
#pragma unroll
  for (int j = 0; j < 61; ++j) { xr[j] = o[j]; hr[j] = (_Float16)o[j]; }
  xr[61] = f0; xr[62] = f1; xr[63] = 1.f;
  hr[61] = (_Float16)f0; hr[62] = (_Float16)f1; hr[63] = (_Float16)1.f;
  if (isVar) {
    _Float16* hc = hcath + (size_t)i * 320;
#pragma unroll
    for (int j = 0; j < 61; ++j) hc[j] = (_Float16)o[j];
    hc[61] = (_Float16)f0; hc[62] = (_Float16)f1; hc[63] = (_Float16)1.f;
  }
}

// ---------------------------------------------------------------- edge WMMA kernel
// One wave per 16-edge tile. A = 16x64 gathered xj (f16); B = 64x64 padded weights,
// staged once per block in LDS (24 KB) so the inner loop's B-fragments come from
// ds_load_b128 and the vmem path is reserved for gathers + atomics.
__global__ void __launch_bounds__(256) k_edge(
    const _Float16* __restrict__ xh, const float* __restrict__ x32,
    const int* __restrict__ src, const int* __restrict__ dstp,
    const int* __restrict__ ety, const float* __restrict__ ef,
    const _Float16* __restrict__ w1t, const _Float16* __restrict__ wct,
    const _Float16* __restrict__ wvt, const float* __restrict__ b1,
    const float* __restrict__ w2, const float* __restrict__ b2p,
    float* __restrict__ aggr, float* __restrict__ deg, int numTiles) {
  __shared__ _Float16 sW[3 * 4096];      // [w1t | wct | wvt] for this layer
  const int tid = threadIdx.x;
  {
#pragma unroll
    for (int mtx = 0; mtx < 3; ++mtx) {
      const _Float16* s = (mtx == 0) ? w1t : (mtx == 1) ? wct : wvt;
#pragma unroll
      for (int i = 0; i < 2; ++i) {
        int off = tid * 8 + i * 2048;
        *reinterpret_cast<v8h*>(sW + mtx * 4096 + off) =
            *reinterpret_cast<const v8h*>(s + off);
      }
    }
  }
  __syncthreads();

  const int lane = tid & 31;
  const int wv = tid >> 5;
  const int wpb = blockDim.x >> 5;
  const int row = lane & 15;
  const int hi = lane >> 4;
  const float b2s = b2p[0];
  const int bOff = (row << 6) + hi * 16;  // lane's B-fragment base within a 64x64 tile row

  // tile-loop-invariant per-lane, per-ntile scalars (avoids divergent loads inside the tile loop)
  float bnv[4], w2v[4];
#pragma unroll
  for (int nt = 0; nt < 4; ++nt) {
    int n = nt * 16 + row;
    bnv[nt] = (n < 63) ? b1[n] : 0.f;
    w2v[nt] = (n < 63) ? w2[n] : 0.f;
  }

  for (int tile = blockIdx.x * wpb + wv; tile < numTiles; tile += gridDim.x * wpb) {
    int e = tile * 16 + row;                 // lanes 16..31 replicate lanes 0..15
    int s = src[e], d = dstp[e], t = ety[e];
    float c = ef[s];
    float xl = x32[(size_t)s * 64 + 63];
    unsigned m0 = (unsigned)__ballot(t == 0);
    bool anyC = (m0 != 0u);
    bool anyV = (m0 != 0xFFFFFFFFu);

    // A fragments for K-steps 0 and 32 (contiguous k = 16*hi + s, matching B)
    const _Float16* rp = xh + (size_t)s * 64;
    V16 A0, A1;
    A0.h[0] = *reinterpret_cast<const v8h*>(rp + hi * 16);
    A0.h[1] = *reinterpret_cast<const v8h*>(rp + hi * 16 + 8);
    A1.h[0] = *reinterpret_cast<const v8h*>(rp + 32 + hi * 16);
    A1.h[1] = *reinterpret_cast<const v8h*>(rp + 32 + hi * 16 + 8);

    float red[8];
#pragma unroll
    for (int r = 0; r < 8; ++r) red[r] = 0.f;

    if (anyC) {  // sigmoid MLP for var_assign (type-0 edges) — wave-uniform branch
      float part[8];
#pragma unroll
      for (int r = 0; r < 8; ++r) part[r] = 0.f;
#pragma unroll
      for (int nt = 0; nt < 4; ++nt) {
        const _Float16* wp = sW + nt * 1024 + bOff;
        V16 B0, B1;
        B0.h[0] = *reinterpret_cast<const v8h*>(wp);
        B0.h[1] = *reinterpret_cast<const v8h*>(wp + 8);
        B1.h[0] = *reinterpret_cast<const v8h*>(wp + 32);
        B1.h[1] = *reinterpret_cast<const v8h*>(wp + 40);
        v8f acc = {};
        acc = wmma32(A0, B0, acc);
        acc = wmma32(A1, B1, acc);
#pragma unroll
        for (int r = 0; r < 8; ++r) {
          float hsv = 1.f / (1.f + __expf(-(acc[r] + bnv[nt])));
          part[r] += hsv * w2v[nt];
        }
      }
#pragma unroll
      for (int off = 8; off > 0; off >>= 1) {
#pragma unroll
        for (int r = 0; r < 8; ++r) part[r] += __shfl_xor(part[r], off);
      }
#pragma unroll
      for (int r = 0; r < 8; ++r) red[r] = part[r] + b2s;  // rows m = r + 8*hi
    }

    // per-row destination / type / column-63 value (r-dependent only; hoisted)
    int dmv[8], tmv[8];
    float v63[8];
#pragma unroll
    for (int r = 0; r < 8; ++r) {
      int m = r + hi * 8;                 // C-frag row held by this lane
      dmv[r] = __shfl(d, m);
      tmv[r] = __shfl(t, m);
      float cm = __shfl(c, m);
      float xlm = __shfl(xl, m);
      v63[r] = (tmv[r] == 0) ? red[r] * cm : cm * xlm;
    }

    // head @ (w_cons | w_vars), scatter messages
#pragma unroll
    for (int nt = 0; nt < 4; ++nt) {
      v8f accC = {}, accV = {};
      if (anyC) {
        const _Float16* wp = sW + 4096 + nt * 1024 + bOff;
        V16 B0, B1;
        B0.h[0] = *reinterpret_cast<const v8h*>(wp);
        B0.h[1] = *reinterpret_cast<const v8h*>(wp + 8);
        B1.h[0] = *reinterpret_cast<const v8h*>(wp + 32);
        B1.h[1] = *reinterpret_cast<const v8h*>(wp + 40);
        accC = wmma32(A0, B0, accC);
        accC = wmma32(A1, B1, accC);
      }
      if (anyV) {
        const _Float16* wp = sW + 8192 + nt * 1024 + bOff;
        V16 B0, B1;
        B0.h[0] = *reinterpret_cast<const v8h*>(wp);
        B0.h[1] = *reinterpret_cast<const v8h*>(wp + 8);
        B1.h[0] = *reinterpret_cast<const v8h*>(wp + 32);
        B1.h[1] = *reinterpret_cast<const v8h*>(wp + 40);
        accV = wmma32(A0, B0, accV);
        accV = wmma32(A1, B1, accV);
      }
      int n = nt * 16 + row;
#pragma unroll
      for (int r = 0; r < 8; ++r) {
        float val = (tmv[r] == 0) ? accC[r] : accV[r];
        if (n == 63) val = v63[r];        // lanes 15/31 hold column 63
        atomicAdd(aggr + (size_t)dmv[r] * 64 + n, val);
      }
    }
    if (hi == 0) atomicAdd(deg + d, 1.f);
  }
}

// scalar fallback for E % 16 tail edges
__global__ void k_edge_rem(const float* __restrict__ x32, const int* __restrict__ src,
                           const int* __restrict__ dstp, const int* __restrict__ ety,
                           const float* __restrict__ ef, const float* __restrict__ w1,
                           const float* __restrict__ b1, const float* __restrict__ w2,
                           const float* __restrict__ b2p, const float* __restrict__ wcons,
                           const float* __restrict__ wvars, float* __restrict__ aggr,
                           float* __restrict__ deg, int e0, int E) {
  int e = e0 + blockIdx.x * blockDim.x + threadIdx.x;
  if (e >= E) return;
  int s = src[e], d = dstp[e], t = ety[e];
  float c = ef[s];
  const float* xr = x32 + (size_t)s * 64;
  float xk[64];
#pragma unroll
  for (int k = 0; k < 64; ++k) xk[k] = xr[k];
  const float* W = (t == 0) ? wcons : wvars;
  for (int n = 0; n < 63; ++n) {
    float a = 0.f;
#pragma unroll
    for (int k = 0; k < 63; ++k) a += xk[k] * W[k * 63 + n];
    atomicAdd(aggr + (size_t)d * 64 + n, a);
  }
  float msg63;
  if (t == 0) {
    float va = b2p[0];
    for (int n = 0; n < 63; ++n) {
      float h = 0.f;
#pragma unroll
      for (int k = 0; k < 64; ++k) h += xk[k] * w1[k * 63 + n];
      h = 1.f / (1.f + __expf(-(h + b1[n])));
      va += h * w2[n];
    }
    msg63 = va * c;
  } else {
    msg63 = c * xk[63];
  }
  atomicAdd(aggr + (size_t)d * 64 + 63, msg63);
  atomicAdd(deg + d, 1.f);
}

// deg-divide, rhs fixup, bias, relu; writes f32 x, f16 mirror, f16 concat slice
__global__ void k_finalize(float* __restrict__ x, _Float16* __restrict__ xh,
                           _Float16* __restrict__ hcath, const float* __restrict__ deg,
                           const int* __restrict__ rowinfo, const float* __restrict__ rhs,
                           const float* __restrict__ bias, int l, int nVar,
                           long long total) {
  long long gid = (long long)blockIdx.x * blockDim.x + threadIdx.x;
  if (gid >= total) return;
  int rowIdx = (int)(gid >> 6);
  int col = (int)(gid & 63);
  int info = rowinfo[rowIdx];
  float v;
  if (info < 0) v = 0.f;
  else {
    float dg = fmaxf(deg[rowIdx], 1.f);
    v = x[gid] / dg;
    if (info >= nVar && col == 63) v -= rhs[info - nVar];
  }
  v += bias[col];
  v = fmaxf(v, 0.f);
  x[gid] = v;
  xh[gid] = (_Float16)v;
  if (info >= 0 && info < nVar)
    hcath[(size_t)info * 320 + (size_t)(l + 1) * 64 + col] = (_Float16)v;
}

// ---------------------------------------------------------------- final head MLP
__global__ void __launch_bounds__(256) k_head(
    const _Float16* __restrict__ hcath, const _Float16* __restrict__ fc1t,
    const _Float16* __restrict__ fc2t, const _Float16* __restrict__ fc3t,
    const float* __restrict__ fc1b, const float* __restrict__ fc2b,
    const float* __restrict__ fc3b, const float* __restrict__ fc4w,
    const float* __restrict__ fc4b, float* __restrict__ out, int nVar, int numTiles) {
  __shared__ _Float16 lds[8 * 1024];
  const int lane = threadIdx.x & 31;
  const int wv = threadIdx.x >> 5;
  const int wpb = blockDim.x >> 5;
  const int row = lane & 15;
  const int hi = lane >> 4;
  _Float16* tl = lds + wv * 1024;        // private 16x64 f16 tile per wave
  v8f zero8 = {};
  for (int tile = blockIdx.x * wpb + wv; tile < numTiles; tile += gridDim.x * wpb) {
    int base = tile * 16;
    int node = base + row;
    if (node >= nVar) node = nVar - 1;
    const _Float16* hp = hcath + (size_t)node * 320;
    v8f acc[4];
#pragma unroll
    for (int nt = 0; nt < 4; ++nt) acc[nt] = zero8;
    // fc1: K=320 -> 10 K-steps
    for (int ks = 0; ks < 10; ++ks) {
      V16 A;
      A.h[0] = *reinterpret_cast<const v8h*>(hp + ks * 32 + hi * 16);
      A.h[1] = *reinterpret_cast<const v8h*>(hp + ks * 32 + hi * 16 + 8);
#pragma unroll
      for (int nt = 0; nt < 4; ++nt) {
        const _Float16* wp = fc1t + (size_t)(nt * 16 + row) * 320 + ks * 32 + hi * 16;
        V16 B;
        B.h[0] = *reinterpret_cast<const v8h*>(wp);
        B.h[1] = *reinterpret_cast<const v8h*>(wp + 8);
        acc[nt] = wmma32(A, B, acc[nt]);
      }
    }
#pragma unroll
    for (int nt = 0; nt < 4; ++nt) {       // bias + relu -> LDS (f16, row-major)
      int n = nt * 16 + row;
      float bn = fc1b[n];
#pragma unroll
      for (int r = 0; r < 8; ++r) {
        int m = r + hi * 8;
        tl[m * 64 + n] = (_Float16)fmaxf(acc[nt][r] + bn, 0.f);
      }
    }
    // fc2
    V16 A0, A1;
    A0.h[0] = *reinterpret_cast<const v8h*>(tl + row * 64 + hi * 16);
    A0.h[1] = *reinterpret_cast<const v8h*>(tl + row * 64 + hi * 16 + 8);
    A1.h[0] = *reinterpret_cast<const v8h*>(tl + row * 64 + 32 + hi * 16);
    A1.h[1] = *reinterpret_cast<const v8h*>(tl + row * 64 + 32 + hi * 16 + 8);
#pragma unroll
    for (int nt = 0; nt < 4; ++nt) {
      const _Float16* wp = fc2t + ((nt * 16 + row) << 6) + hi * 16;
      V16 B0, B1;
      B0.h[0] = *reinterpret_cast<const v8h*>(wp);
      B0.h[1] = *reinterpret_cast<const v8h*>(wp + 8);
      B1.h[0] = *reinterpret_cast<const v8h*>(wp + 32);
      B1.h[1] = *reinterpret_cast<const v8h*>(wp + 40);
      v8f a = zero8;
      a = wmma32(A0, B0, a);
      a = wmma32(A1, B1, a);
      acc[nt] = a;
    }
#pragma unroll
    for (int nt = 0; nt < 4; ++nt) {
      int n = nt * 16 + row;
      float bn = fc2b[n];
#pragma unroll
      for (int r = 0; r < 8; ++r) {
        int m = r + hi * 8;
        tl[m * 64 + n] = (_Float16)fmaxf(acc[nt][r] + bn, 0.f);
      }
    }
    // fc3
    A0.h[0] = *reinterpret_cast<const v8h*>(tl + row * 64 + hi * 16);
    A0.h[1] = *reinterpret_cast<const v8h*>(tl + row * 64 + hi * 16 + 8);
    A1.h[0] = *reinterpret_cast<const v8h*>(tl + row * 64 + 32 + hi * 16);
    A1.h[1] = *reinterpret_cast<const v8h*>(tl + row * 64 + 32 + hi * 16 + 8);
#pragma unroll
    for (int nt = 0; nt < 4; ++nt) {
      const _Float16* wp = fc3t + ((nt * 16 + row) << 6) + hi * 16;
      V16 B0, B1;
      B0.h[0] = *reinterpret_cast<const v8h*>(wp);
      B0.h[1] = *reinterpret_cast<const v8h*>(wp + 8);
      B1.h[0] = *reinterpret_cast<const v8h*>(wp + 32);
      B1.h[1] = *reinterpret_cast<const v8h*>(wp + 40);
      v8f a = zero8;
      a = wmma32(A0, B0, a);
      a = wmma32(A1, B1, a);
      acc[nt] = a;
    }
    // fc4: relu(h3) . fc4w  (VALU dot + butterfly reduce within 16-lane halves)
    float part[8];
#pragma unroll
    for (int r = 0; r < 8; ++r) part[r] = 0.f;
#pragma unroll
    for (int nt = 0; nt < 4; ++nt) {
      int n = nt * 16 + row;
      float bn = fc3b[n];
      float w4 = fc4w[n];
#pragma unroll
      for (int r = 0; r < 8; ++r) part[r] += fmaxf(acc[nt][r] + bn, 0.f) * w4;
    }
#pragma unroll
    for (int off = 8; off > 0; off >>= 1) {
#pragma unroll
      for (int r = 0; r < 8; ++r) part[r] += __shfl_xor(part[r], off);
    }
    if (row == 0) {
      float b4 = fc4b[0];
#pragma unroll
      for (int r = 0; r < 8; ++r) {
        int nd = base + r + hi * 8;
        if (nd < nVar) out[nd] = part[r] + b4;
      }
    }
  }
}

// ---------------------------------------------------------------- host
extern "C" void kernel_launch(void* const* d_in, const int* in_sizes, int n_in,
                              void* d_out, int out_size, void* d_ws, size_t ws_size,
                              hipStream_t stream) {
  (void)n_in; (void)out_size; (void)ws_size;
  const float* vnf = (const float*)d_in[0];
  const float* cnf = (const float*)d_in[1];
  const float* ef  = (const float*)d_in[2];
  const float* rhs = (const float*)d_in[3];
  const int* eidx  = (const int*)d_in[4];
  const int* ety   = (const int*)d_in[5];
  const int* av    = (const int*)d_in[6];
  const int* ac    = (const int*)d_in[7];
  const float* vw1 = (const float*)d_in[8];
  const float* vb1 = (const float*)d_in[9];
  const float* vw2 = (const float*)d_in[10];
  const float* vb2 = (const float*)d_in[11];
  const float* cw1 = (const float*)d_in[12];
  const float* cb1 = (const float*)d_in[13];
  const float* cw2 = (const float*)d_in[14];
  const float* cb2 = (const float*)d_in[15];
  const float* h2v_w1 = (const float*)d_in[16];
  const float* h2v_b1 = (const float*)d_in[17];
  const float* h2v_w2 = (const float*)d_in[18];
  const float* h2v_b2 = (const float*)d_in[19];
  const float* w_cons = (const float*)d_in[20];
  const float* w_vars = (const float*)d_in[21];
  const float* conv_bias = (const float*)d_in[22];
  const float* fc1w = (const float*)d_in[23];
  const float* fc1b = (const float*)d_in[24];
  const float* fc2w = (const float*)d_in[25];
  const float* fc2b = (const float*)d_in[26];
  const float* fc3w = (const float*)d_in[27];
  const float* fc3b = (const float*)d_in[28];
  const float* fc4w = (const float*)d_in[29];
  const float* fc4b = (const float*)d_in[30];

  int nVar = in_sizes[0] / 2;
  int nCon = in_sizes[1] / 2;
  int N = nVar + nCon;
  int E = in_sizes[5];
  int L = in_sizes[19];
  const int* src = eidx;
  const int* dstp = eidx + E;

  char* p = (char*)d_ws;
  auto carve = [&](size_t bytes) -> char* {
    char* r = p;
    p += (bytes + 255) & ~(size_t)255;
    return r;
  };
  float* xA = (float*)carve((size_t)N * 64 * 4);
  float* xB = (float*)carve((size_t)N * 64 * 4);
  _Float16* xh = (_Float16*)carve((size_t)N * 64 * 2);
  float* deg = (float*)carve((size_t)N * 4);
  _Float16* hcath = (_Float16*)carve((size_t)nVar * 320 * 2);
  int* rowinfo = (int*)carve((size_t)N * 4);
  _Float16* w1t = (_Float16*)carve((size_t)L * 4096 * 2);
  _Float16* wct = (_Float16*)carve((size_t)L * 4096 * 2);
  _Float16* wvt = (_Float16*)carve((size_t)L * 4096 * 2);
  _Float16* fc1t = (_Float16*)carve((size_t)64 * 320 * 2);
  _Float16* fc2t = (_Float16*)carve((size_t)4096 * 2);
  _Float16* fc3t = (_Float16*)carve((size_t)4096 * 2);

  int wtot = L * 4096 * 3 + 64 * 320 + 4096 + 4096;
  k_wprep<<<(wtot + 255) / 256, 256, 0, stream>>>(h2v_w1, w_cons, w_vars, fc1w, fc2w, fc3w,
                                                  w1t, wct, wvt, fc1t, fc2t, fc3t, L);
  k_rowinit<<<(N + 255) / 256, 256, 0, stream>>>(rowinfo, N);
  k_rowscatter<<<(N + 255) / 256, 256, 0, stream>>>(rowinfo, av, ac, nVar, nCon);
  k_zero<<<(int)(((long long)N * 64 + 255) / 256), 256, 0, stream>>>(xA, (long long)N * 64);
  k_zero<<<(int)(((long long)N * 32 + 255) / 256), 256, 0, stream>>>((float*)xh,
                                                                     (long long)N * 32);
  k_embed<<<(N + 255) / 256, 256, 0, stream>>>(vnf, cnf, vw1, vb1, vw2, vb2, cw1, cb1, cw2,
                                               cb2, av, ac, xA, xh, hcath, nVar, nCon);

  float* cur = xA;
  float* nxt = xB;
  int numTiles = E / 16;
  int rem = E % 16;
  for (int l = 0; l < L; ++l) {
    k_zero<<<(int)(((long long)N * 64 + 255) / 256), 256, 0, stream>>>(nxt, (long long)N * 64);
    k_zero<<<(N + 255) / 256, 256, 0, stream>>>(deg, N);
    if (numTiles > 0) {
      int blocks = (numTiles + 7) / 8;
      if (blocks > 2048) blocks = 2048;   // amortize LDS weight staging over ~4 tiles/wave
      k_edge<<<blocks, 256, 0, stream>>>(xh, cur, src, dstp, ety, ef,
                                         w1t + (size_t)l * 4096, wct + (size_t)l * 4096,
                                         wvt + (size_t)l * 4096, h2v_b1 + l * 63,
                                         h2v_w2 + l * 63, h2v_b2 + l, nxt, deg, numTiles);
    }
    if (rem > 0) {
      k_edge_rem<<<1, 32, 0, stream>>>(cur, src, dstp, ety, ef, h2v_w1 + (size_t)l * 64 * 63,
                                       h2v_b1 + l * 63, h2v_w2 + l * 63, h2v_b2 + l,
                                       w_cons + (size_t)l * 63 * 63,
                                       w_vars + (size_t)l * 63 * 63, nxt, deg, numTiles * 16,
                                       E);
    }
    k_finalize<<<(int)(((long long)N * 64 + 255) / 256), 256, 0, stream>>>(
        nxt, xh, hcath, deg, rowinfo, rhs, conv_bias + l * 64, l, nVar, (long long)N * 64);
    float* tmp = cur; cur = nxt; nxt = tmp;
  }

  int hTiles = (nVar + 15) / 16;
  k_head<<<(hTiles + 7) / 8, 256, 0, stream>>>(hcath, fc1t, fc2t, fc3t, fc1b, fc2b, fc3b,
                                               fc4w, fc4b, (float*)d_out, nVar, hTiles);
}
